// PerformerAttention_3418793968263
// MI455X (gfx1250) — matrix operand order repaired and checked by reading, not compiled, not run on hardware
//
#include <hip/hip_runtime.h>
#include <hip/hip_bf16.h>
#include <stdint.h>

#define Bc 4
#define Hc 16
#define Nc 8192
#define Dc 1024
#define Fc 64
#define EPSf 1e-8f

typedef __attribute__((ext_vector_type(16))) __bf16    v16bf;
typedef __attribute__((ext_vector_type(8)))  __bf16    v8bf;
typedef __attribute__((ext_vector_type(2)))  __bf16    v2bf;
typedef __attribute__((ext_vector_type(8)))  float     v8f;
typedef __attribute__((ext_vector_type(4)))  float     v4f;
typedef __attribute__((ext_vector_type(4)))  uint32_t  v4u;
typedef __attribute__((ext_vector_type(8)))  uint32_t  v8u;
typedef __attribute__((ext_vector_type(4)))  int       v4i;

#if __has_builtin(__builtin_amdgcn_global_load_async_to_lds_b128) && \
    __has_builtin(__builtin_amdgcn_s_wait_asynccnt)
#define HAVE_ASYNC_LDS 1
#endif

__device__ __forceinline__ v8f wmma_bf16(v16bf a, v16bf b, v8f c) {
    return __builtin_amdgcn_wmma_f32_16x16x32_bf16(false, a, false, b, (short)0, c, false, false);
}

// 16x32 bf16 fragment from two 16-byte-aligned runs of 8 contiguous bf16.
__device__ __forceinline__ v16bf frag_from(const __bf16* p0, const __bf16* p1) {
    v4u a = *(const v4u*)p0;
    v4u b = *(const v4u*)p1;
    v8u u;
    u[0] = a[0]; u[1] = a[1]; u[2] = a[2]; u[3] = a[3];
    u[4] = b[0]; u[5] = b[1]; u[6] = b[2]; u[7] = b[3];
    return __builtin_bit_cast(v16bf, u);
}

__device__ __forceinline__ uint32_t pk2(float lo, float hi) {
    union { __bf16 b[2]; uint32_t u; } p;
    p.b[0] = (__bf16)lo; p.b[1] = (__bf16)hi;
    return p.u;
}
__device__ __forceinline__ uint32_t pk2b(__bf16 lo, __bf16 hi) {
    union { __bf16 b[2]; uint32_t u; } p;
    p.b[0] = lo; p.b[1] = hi;
    return p.u;
}

// -------------------------------------------------------------------------
// Stage 1: phi = L2-normalized exp(X @ FM[h]).  grid (N/128, B*H), 256 thr.
// -------------------------------------------------------------------------
__global__ __launch_bounds__(256) void k_featurize(
    const float* __restrict__ X,     // (B, N, D)
    const float* __restrict__ FM,    // (H, 64, 64)  [d][f]
    __bf16* __restrict__ phi)        // (B*H, N, 64) bf16
{
    const int bh   = blockIdx.y;
    const int b    = bh / Hc, h = bh % Hc;
    const int row0 = blockIdx.x * 128;
    const int tid  = threadIdx.x;
    const int wave = tid >> 5, lane = tid & 31;
    const int mrow = lane & 15, half = lane >> 4;
    const int wrow = wave * 16;

    __shared__ __bf16 sX[128 * 64];    // 16 KB, row-major [row][d]
    __shared__ __bf16 sFt[64 * 72];    //  9 KB, transposed [f][d], stride 72

    // FM transpose fill, d-pairs packed into u32 stores
    for (int i = tid; i < 512; i += 256) {
        int d2 = (i >> 4) * 2, f0 = (i & 15) * 4;
        v4f x0 = *(const v4f*)&FM[(size_t)h * 4096 + d2 * 64 + f0];
        v4f x1 = *(const v4f*)&FM[(size_t)h * 4096 + (d2 + 1) * 64 + f0];
#pragma unroll
        for (int j = 0; j < 4; ++j)
            *(uint32_t*)&sFt[(f0 + j) * 72 + d2] = pk2(x0[j], x1[j]);
    }
    // X fill: float4 reads -> packed 8-byte LDS stores
    for (int i = tid; i < 2048; i += 256) {
        int r = i >> 4, c0 = (i & 15) * 4;
        v4f x = *(const v4f*)&X[((size_t)b * Nc + row0 + r) * Dc + h * 64 + c0];
        uint2 u;
        u.x = pk2(x[0], x[1]);
        u.y = pk2(x[2], x[3]);
        *(uint2*)&sX[r * 64 + c0] = u;
    }
    __syncthreads();

    const __bf16* arow = &sX[(wrow + mrow) * 64];
    v16bf a0 = frag_from(arow + half * 8,      arow + 16 + half * 8);
    v16bf a1 = frag_from(arow + 32 + half * 8, arow + 48 + half * 8);

    // all 8 WMMAs into 4 independent accumulators first (hides D->VALU hazard)
    v8f accv[4];
#pragma unroll
    for (int ft = 0; ft < 4; ++ft) {
        const __bf16* bcol = &sFt[(ft * 16 + mrow) * 72];
        v8f acc = {};
        acc = wmma_bf16(a0, frag_from(bcol + half * 8,      bcol + 16 + half * 8), acc);
        acc = wmma_bf16(a1, frag_from(bcol + 32 + half * 8, bcol + 48 + half * 8), acc);
        accv[ft] = acc;
    }
    float p[4][8];
#pragma unroll
    for (int ft = 0; ft < 4; ++ft)
#pragma unroll
        for (int r = 0; r < 8; ++r) p[ft][r] = __expf(accv[ft][r]);

    // row sum-of-squares across the 16 lanes of each half (xor 1,2,4,8)
    float rinv[8];
#pragma unroll
    for (int r = 0; r < 8; ++r) {
        float s = 0.f;
#pragma unroll
        for (int ft = 0; ft < 4; ++ft) s += p[ft][r] * p[ft][r];
#pragma unroll
        for (int off = 1; off < 16; off <<= 1) s += __shfl_xor(s, off, 32);
        rinv[r] = 1.f / (sqrtf(s) + EPSf);
    }

    // restage normalized phi into sX, then coalesced b128 global stores
    __syncthreads();
#pragma unroll
    for (int ft = 0; ft < 4; ++ft)
#pragma unroll
        for (int r = 0; r < 8; ++r)
            sX[(wrow + r + half * 8) * 64 + ft * 16 + mrow] = (__bf16)(p[ft][r] * rinv[r]);
    __syncthreads();

    v4u* dst = (v4u*)&phi[((size_t)bh * Nc + row0) * 64];
    const v4u* src = (const v4u*)sX;
    for (int i = tid; i < 1024; i += 256) dst[i] = src[i];
}

// -------------------------------------------------------------------------
// Stage 2: KV[f][v] += phi_K^T @ V over 256-row chunks (fp32 atomics).
// grid (N/256, B*H), 256 thr.
// -------------------------------------------------------------------------
__global__ __launch_bounds__(256) void k_kv(
    const __bf16* __restrict__ phiK,  // (B*H, N, 64)
    const float*  __restrict__ V,     // (B, N, D)
    float* __restrict__ KV)           // (B*H, 64, 64) [f][v]
{
    const int bh = blockIdx.y;
    const int b  = bh / Hc, h = bh % Hc;
    const int n0 = blockIdx.x * 256;
    const int tid = threadIdx.x, wave = tid >> 5, lane = tid & 31;
    const int mrow = lane & 15, half = lane >> 4;

    __shared__ __bf16 sKt[64 * 264];   // 33 KB, transposed [f][n], stride 264
    __shared__ __bf16 sVt[64 * 264];   // 33 KB, transposed [v][n], stride 264

    // phiK transpose: n-pairs packed into u32 stores
    for (int i = tid; i < 1024; i += 256) {
        int n2 = (i >> 3) * 2, f0 = (i & 7) * 8;
        v8bf p0 = *(const v8bf*)&phiK[((size_t)bh * Nc + n0 + n2) * 64 + f0];
        v8bf p1 = *(const v8bf*)&phiK[((size_t)bh * Nc + n0 + n2 + 1) * 64 + f0];
#pragma unroll
        for (int j = 0; j < 8; ++j)
            *(uint32_t*)&sKt[(f0 + j) * 264 + n2] = pk2b(p0[j], p1[j]);
    }
    // V transpose: n-pairs packed into u32 stores
    for (int i = tid; i < 2048; i += 256) {
        int n2 = (i >> 4) * 2, v0 = (i & 15) * 4;
        v4f x0 = *(const v4f*)&V[((size_t)b * Nc + n0 + n2) * Dc + h * 64 + v0];
        v4f x1 = *(const v4f*)&V[((size_t)b * Nc + n0 + n2 + 1) * Dc + h * 64 + v0];
#pragma unroll
        for (int j = 0; j < 4; ++j)
            *(uint32_t*)&sVt[(v0 + j) * 264 + n2] = pk2(x0[j], x1[j]);
    }
    __syncthreads();

    float* dst = KV + (size_t)bh * 64 * 64;
    for (int t = wave; t < 16; t += 8) {
        const int ftile = t >> 2, vtile = t & 3;
        const __bf16* arow = &sKt[(ftile * 16 + mrow) * 264];
        const __bf16* brow = &sVt[(vtile * 16 + mrow) * 264];
        v8f acc = {};
#pragma unroll
        for (int kk = 0; kk < 256; kk += 32) {
            v16bf a  = frag_from(arow + kk + half * 8,  arow + kk + 16 + half * 8);
            v16bf bb = frag_from(brow + kk + half * 8, brow + kk + 16 + half * 8);
            acc = wmma_bf16(a, bb, acc);
        }
#pragma unroll
        for (int r = 0; r < 8; ++r)
            atomicAdd(dst + (ftile * 16 + r + half * 8) * 64 + vtile * 16 + mrow, acc[r]);
    }
}

// -------------------------------------------------------------------------
// Stage 3: out = (phi_Q @ KV) / (<phi_Q,phi_K> + eps).  grid (N/128, B*H).
// -------------------------------------------------------------------------
__global__ __launch_bounds__(256) void k_out(
    const __bf16* __restrict__ phiQ,
    const __bf16* __restrict__ phiK,
    const float*  __restrict__ KV,
    float* __restrict__ out,          // (B, N, D)
    float* __restrict__ pqpk)         // (B*H, N)
{
    const int bh   = blockIdx.y;
    const int b    = bh / Hc, h = bh % Hc;
    const int row0 = blockIdx.x * 128;
    const int tid  = threadIdx.x, wave = tid >> 5, lane = tid & 31;
    const int mrow = lane & 15, half = lane >> 4;
    const int wrow = wave * 16;

    __shared__ __bf16 sQ[128 * 64];    // 16 KB, row-major [row][f]
    __shared__ __bf16 sKVt[64 * 72];   //  9 KB, transposed [v][f], stride 72
    __shared__ float  sDen[128];

#ifdef HAVE_ASYNC_LDS
    {   // async DMA copy global -> LDS (tracked by ASYNCcnt)
        const __bf16* gsrc = &phiQ[((size_t)bh * Nc + row0) * 64];
        for (int i = tid; i < 1024; i += 256)
            __builtin_amdgcn_global_load_async_to_lds_b128(
                (__attribute__((address_space(1))) v4i*)(gsrc + (size_t)i * 8),
                (__attribute__((address_space(3))) v4i*)(sQ + (size_t)i * 8),
                0, 0);
    }
#else
    {   // fallback: b128 copy through VGPRs
        v4u* d = (v4u*)sQ;
        const v4u* s = (const v4u*)&phiQ[((size_t)bh * Nc + row0) * 64];
        for (int i = tid; i < 1024; i += 256) d[i] = s[i];
    }
#endif
    // KV fp32 -> bf16, transposed, f-pairs packed into u32 stores
    for (int i = tid; i < 512; i += 256) {
        int f2 = (i >> 4) * 2, v0 = (i & 15) * 4;
        v4f x0 = *(const v4f*)&KV[(size_t)bh * 4096 + f2 * 64 + v0];
        v4f x1 = *(const v4f*)&KV[(size_t)bh * 4096 + (f2 + 1) * 64 + v0];
#pragma unroll
        for (int j = 0; j < 4; ++j)
            *(uint32_t*)&sKVt[(v0 + j) * 72 + f2] = pk2(x0[j], x1[j]);
    }
#ifdef HAVE_ASYNC_LDS
    __builtin_amdgcn_s_wait_asynccnt(0);
#endif
    __syncthreads();

    if (tid < 128) {
        const v8bf* qv = (const v8bf*)&sQ[tid * 64];
        const v8bf* kv = (const v8bf*)&phiK[((size_t)bh * Nc + row0 + tid) * 64];
        float s = 0.f;
#pragma unroll
        for (int j = 0; j < 8; ++j) {
#if __has_builtin(__builtin_amdgcn_fdot2_f32_bf16)
            v4u uq = __builtin_bit_cast(v4u, qv[j]);
            v4u uk = __builtin_bit_cast(v4u, kv[j]);
#pragma unroll
            for (int e = 0; e < 4; ++e)
                s = __builtin_amdgcn_fdot2_f32_bf16(
                        __builtin_bit_cast(v2bf, uq[e]),
                        __builtin_bit_cast(v2bf, uk[e]), s, false);
#else
            v8bf q = qv[j], k2 = kv[j];
#pragma unroll
            for (int e = 0; e < 8; ++e) s += (float)q[e] * (float)k2[e];
#endif
        }
        sDen[tid] = s + EPSf;
        pqpk[(size_t)bh * Nc + row0 + tid] = s;
    }
    __syncthreads();

    const __bf16* arow = &sQ[(wrow + mrow) * 64];
    v16bf a0 = frag_from(arow + half * 8,      arow + 16 + half * 8);
    v16bf a1 = frag_from(arow + 32 + half * 8, arow + 48 + half * 8);

#pragma unroll
    for (int vt = 0; vt < 4; ++vt) {
        const __bf16* bcol = &sKVt[(vt * 16 + mrow) * 72];
        v8f acc = {};
        acc = wmma_bf16(a0, frag_from(bcol + half * 8,      bcol + 16 + half * 8), acc);
        acc = wmma_bf16(a1, frag_from(bcol + 32 + half * 8, bcol + 48 + half * 8), acc);
#pragma unroll
        for (int r = 0; r < 8; ++r) {
            int row = wrow + r + half * 8;
            out[((size_t)b * Nc + row0 + row) * Dc + h * 64 + vt * 16 + mrow] =
                acc[r] / sDen[row];
        }
    }
}

// -------------------------------------------------------------------------
// Stage 4: attn_weights = pq_pk / (sum_n pq_pk + eps).  grid B*H, 256 thr.
// -------------------------------------------------------------------------
__global__ __launch_bounds__(256) void k_attn(
    const float* __restrict__ pqpk, float* __restrict__ attn)
{
    const int bh = blockIdx.x, tid = threadIdx.x;
    __shared__ float red[256];
    float s = 0.f;
    for (int i = tid; i < Nc; i += 256) s += pqpk[(size_t)bh * Nc + i];
    red[tid] = s;
    __syncthreads();
    for (int o = 128; o > 0; o >>= 1) {
        if (tid < o) red[tid] += red[tid + o];
        __syncthreads();
    }
    const float inv = 1.f / (red[0] + EPSf);
    for (int i = tid; i < Nc; i += 256)
        attn[(size_t)bh * Nc + i] = pqpk[(size_t)bh * Nc + i] * inv;
}

__global__ void k_zero(float* __restrict__ p, int n) {
    int i = blockIdx.x * 256 + threadIdx.x;
    if (i < n) p[i] = 0.f;
}

// -------------------------------------------------------------------------
extern "C" void kernel_launch(void* const* d_in, const int* in_sizes, int n_in,
                              void* d_out, int out_size, void* d_ws, size_t ws_size,
                              hipStream_t stream) {
    const float* q  = (const float*)d_in[0];
    const float* k  = (const float*)d_in[1];
    const float* v  = (const float*)d_in[2];
    const float* fm = (const float*)d_in[3];

    float* out  = (float*)d_out;
    float* attn = out + (size_t)Bc * Nc * Dc;

    char* ws = (char*)d_ws;
    const size_t PHI_BYTES = (size_t)Bc * Hc * Nc * Fc * sizeof(unsigned short);
    __bf16* phiQ = (__bf16*)ws;
    __bf16* phiK = (__bf16*)(ws + PHI_BYTES);
    float*  KV   = (float*)(ws + 2 * PHI_BYTES);
    float*  pqpk = (float*)(ws + 2 * PHI_BYTES + (size_t)Bc * Hc * 64 * 64 * sizeof(float));

    dim3 gfeat(Nc / 128, Bc * Hc);
    k_featurize<<<gfeat, 256, 0, stream>>>(q, fm, phiQ);
    k_featurize<<<gfeat, 256, 0, stream>>>(k, fm, phiK);

    const int kvN = Bc * Hc * 64 * 64;
    k_zero<<<(kvN + 255) / 256, 256, 0, stream>>>(KV, kvN);

    k_kv<<<dim3(Nc / 256, Bc * Hc), 256, 0, stream>>>(phiK, v, KV);
    k_out<<<dim3(Nc / 128, Bc * Hc), 256, 0, stream>>>(phiQ, phiK, KV, out, pqpk);
    k_attn<<<Bc * Hc, 256, 0, stream>>>(pqpk, attn);
}
